// CorrelationSeq2Seq_25245817766426
// MI455X (gfx1250) — matrix operand order repaired and checked by reading, not compiled, not run on hardware
//
#include <hip/hip_runtime.h>

typedef _Float16 f16x8  __attribute__((ext_vector_type(8)));
typedef _Float16 f16x16 __attribute__((ext_vector_type(16)));
typedef float    f32x8  __attribute__((ext_vector_type(8)));

// ---------------------------------------------------------------- helpers

__device__ __forceinline__ f16x16 mk16(f16x8 lo, f16x8 hv) {
    f16x16 r;
#pragma unroll
    for (int i = 0; i < 8; ++i) { r[i] = lo[i]; r[i + 8] = hv[i]; }
    return r;
}

__device__ __forceinline__ f32x8 wmma_f16(f16x16 a, f16x16 b, f32x8 c) {
    return __builtin_amdgcn_wmma_f32_16x16x32_f16(false, a, false, b,
                                                  (short)0, c, false, false);
}

__device__ __forceinline__ float sigf(float x) { return 1.f / (1.f + __expf(-x)); }
__device__ __forceinline__ float lrelu(float x) { return x > 0.f ? x : 0.1f * x; }

// ---------------------------------------------------------------- constants
#define NPAIR 160          // kept correlation pairs (16 tracks * 10)
#define HW    1008         // 24*42
#define NTP   48           // spatial positions per conv block (1008 = 21*48)
#define NTILE 3            // 48/16 N-tiles per wave

// ---------------------------------------------------------------- weight fp32->fp16 (K padded with zeros)
__global__ void wcvt_kernel(const float* __restrict__ src, _Float16* __restrict__ dst,
                            int K, int Kpad, int total) {
    int gid = blockIdx.x * 256 + threadIdx.x;
    if (gid >= total) return;
    int o = gid / Kpad, k = gid - o * Kpad;
    dst[gid] = (_Float16)(k < K ? src[o * K + k] : 0.f);
}

// ---------------------------------------------------------------- correlation (81 shifts, /256, leaky 0.1) -> f16
__global__ void corr_kernel(const float* __restrict__ feat, _Float16* __restrict__ out) {
    int gid = blockIdx.x * 256 + threadIdx.x;          // [i][d][p], 160*81*1008
    int p = gid % HW;
    int rest = gid / HW;
    int d = rest % 81;
    int i = rest / 81;
    int fr = i + i / 10;                               // frame index (skip track bounds)
    int dy = d / 9 - 4, dx = d % 9 - 4;
    int h = p / 42, w = p - h * 42;
    int h2 = h + dy, w2 = w + dx;
    float s = 0.f;
    if (h2 >= 0 && h2 < 24 && w2 >= 0 && w2 < 42) {
        const float* f1 = feat + (long)fr * 256 * HW + h * 42 + w;
        const float* f2 = feat + (long)(fr + 1) * 256 * HW + h2 * 42 + w2;
        for (int c = 0; c < 256; ++c) s += f1[c * HW] * f2[c * HW];
    }
    out[gid] = (_Float16)lrelu(s * (1.f / 256.f));
}

// ---------------------------------------------------------------- implicit-GEMM 3x3 conv (pad 1) via WMMA f16
// one wave per 16-Cout tile x 48 positions; ping-pong im2col tile in LDS,
// 3 WMMAs per chunk reusing one A (weight) fragment; single barrier per chunk.
__global__ void conv_wmma_kernel(const _Float16* __restrict__ in,
                                 const _Float16* __restrict__ w16,  // [Cout][Kpad] f16
                                 const float* __restrict__ bias,
                                 _Float16* __restrict__ out,
                                 int Cin, int Cout, int Kpad) {
    __shared__ __attribute__((aligned(16))) _Float16 Bt[2][NTP * 32]; // [pos 48][k 32]
    const int n = blockIdx.y;
    const int ptile = blockIdx.x * NTP;
    const int tid = threadIdx.x;
    const int wave = tid >> 5, lane = tid & 31;
    const int row = lane & 15, hi = lane >> 4;
    const int Kreal = Cin * 9;
    const _Float16* inb = in + (long)n * Cin * HW;

    // cooperative im2col of one 48x32 chunk into Bt[buf]
    auto fill = [&](int buf, int kk) {
        for (int idx = tid; idx < NTP * 32; idx += blockDim.x) {
            int pos = idx >> 5, kl = idx & 31;
            int k = kk + kl;
            float v = 0.f;
            if (k < Kreal) {
                int ci = k / 9, r = k - ci * 9;
                int ky = r / 3, kx = r - ky * 3;
                int p = ptile + pos;
                int h = p / 42, w = p - h * 42;
                int ih = h + ky - 1, iw = w + kx - 1;
                if (ih >= 0 && ih < 24 && iw >= 0 && iw < 42)
                    v = (float)inb[ci * HW + ih * 42 + iw];
            }
            Bt[buf][pos * 32 + kl] = (_Float16)v;
        }
    };

    f32x8 acc[NTILE];
    f32x8 z = {0.f, 0.f, 0.f, 0.f, 0.f, 0.f, 0.f, 0.f};
#pragma unroll
    for (int t = 0; t < NTILE; ++t) acc[t] = z;

    const int nch = Kpad >> 5;
    fill(0, 0);
    __syncthreads();
    for (int ck = 0; ck < nch; ++ck) {
        const int cur = ck & 1;
        // A fragment: weights, two aligned b128 global loads per lane
        const _Float16* wp = w16 + (wave * 16 + row) * Kpad + (ck << 5);
        f16x16 A = mk16(*(const f16x8*)(wp + 8 * hi), *(const f16x8*)(wp + 16 + 8 * hi));
#pragma unroll
        for (int t = 0; t < NTILE; ++t) {
            const _Float16* bp = &Bt[cur][(t * 16 + row) * 32 + 8 * hi];
            f16x16 B = mk16(*(const f16x8*)bp, *(const f16x8*)(bp + 16));
            acc[t] = wmma_f16(A, B, acc[t]);
        }
        if (ck + 1 < nch) fill(1 - cur, (ck + 1) << 5);
        __syncthreads();
    }

#pragma unroll
    for (int t = 0; t < NTILE; ++t) {
        int p = ptile + t * 16 + row;  // D col = lane%16 -> position within tile
#pragma unroll
        for (int v = 0; v < 8; ++v) {
            int m = wave * 16 + v + 8 * hi;  // D row = v + 8*hi -> Cout
            float y = lrelu(acc[t][v] + bias[m]);
            out[((long)n * Cout + m) * HW + p] = (_Float16)y;
        }
    }
}

// ---------------------------------------------------------------- ROI align (torchvision aligned=False semantics)
__global__ void roi_kernel(const _Float16* __restrict__ feat,   // [160][64][24][42] f16
                           const float* __restrict__ boxes,     // [176][4]
                           const int* __restrict__ image_sizes,
                           float* __restrict__ bf) {            // [160][1024]
    int gid = blockIdx.x * 256 + threadIdx.x;       // i*1024 + c*16 + oy*4 + ox
    int ox = gid & 3, oy = (gid >> 2) & 3, c = (gid >> 4) & 63, i = gid >> 10;
    int fr = i + i / 10;
    float scale = 24.f / (float)image_sizes[0];
    float x1 = boxes[fr * 4 + 0] * scale, y1 = boxes[fr * 4 + 1] * scale;
    float x2 = boxes[fr * 4 + 2] * scale, y2 = boxes[fr * 4 + 3] * scale;
    float bw = fmaxf(x2 - x1, 1.f) * 0.25f;
    float bh = fmaxf(y2 - y1, 1.f) * 0.25f;
    const _Float16* f = feat + ((long)i * 64 + c) * HW;
    float accv = 0.f;
#pragma unroll
    for (int sy = 0; sy < 2; ++sy)
#pragma unroll
        for (int sx = 0; sx < 2; ++sx) {
            float y = y1 + ((float)oy + (sy + 0.5f) * 0.5f) * bh;
            float x = x1 + ((float)ox + (sx + 0.5f) * 0.5f) * bw;
            bool valid = (y >= -1.f) && (y <= 24.f) && (x >= -1.f) && (x <= 42.f);
            y = fmaxf(y, 0.f); x = fmaxf(x, 0.f);
            int y0 = min((int)floorf(y), 23);
            int x0 = min((int)floorf(x), 41);
            int y1i = min(y0 + 1, 23);
            int x1i = min(x0 + 1, 41);
            if (y0 >= 23) y = (float)y0;
            if (x0 >= 41) x = (float)x0;
            float ly = y - (float)y0, lx = x - (float)x0;
            float hy = 1.f - ly, hx = 1.f - lx;
            float v = hy * hx * (float)f[y0 * 42 + x0] + hy * lx * (float)f[y0 * 42 + x1i]
                    + ly * hx * (float)f[y1i * 42 + x0] + ly * lx * (float)f[y1i * 42 + x1i];
            accv += valid ? v : 0.f;
        }
    bf[gid] = accv * 0.25f;
}

// ---------------------------------------------------------------- LSTM gate GEMM helper (WMMA)
// wave owns hidden units [32*wave, 32*wave+32): tiles (gate g, subtile s) -> col base g*256+wave*32+s*16
__device__ __forceinline__ void gate_gemm(const _Float16* __restrict__ x, int Kx,
                                          const _Float16* __restrict__ Wx,
                                          const _Float16* __restrict__ h,
                                          const _Float16* __restrict__ Wh,
                                          int wave, int row, int hi,
                                          f32x8 acc[4][2]) {
    const _Float16* xr = x + row * 1312;
    for (int kk = 0; kk < Kx; kk += 32) {
        f16x16 A = mk16(*(const f16x8*)(xr + kk + 8 * hi),
                        *(const f16x8*)(xr + kk + 16 + 8 * hi));
#pragma unroll
        for (int g = 0; g < 4; ++g)
#pragma unroll
            for (int s = 0; s < 2; ++s) {
                int ncol = g * 256 + wave * 32 + s * 16 + row;
                const _Float16* bp = Wx + ncol * Kx + kk + 8 * hi;
                f16x16 B = mk16(*(const f16x8*)bp, *(const f16x8*)(bp + 16));
                acc[g][s] = wmma_f16(A, B, acc[g][s]);
            }
    }
    const _Float16* hr = h + row * 256;
    for (int kk = 0; kk < 256; kk += 32) {
        f16x16 A = mk16(*(const f16x8*)(hr + kk + 8 * hi),
                        *(const f16x8*)(hr + kk + 16 + 8 * hi));
#pragma unroll
        for (int g = 0; g < 4; ++g)
#pragma unroll
            for (int s = 0; s < 2; ++s) {
                int ncol = g * 256 + wave * 32 + s * 16 + row;
                const _Float16* bp = Wh + ncol * 256 + kk + 8 * hi;
                f16x16 B = mk16(*(const f16x8*)bp, *(const f16x8*)(bp + 16));
                acc[g][s] = wmma_f16(A, B, acc[g][s]);
            }
    }
}

__device__ __forceinline__ void lstm_combine(f32x8 acc[4][2], const float bs[4][2],
                                             float creg[2][8], _Float16* hbuf,
                                             int wave, int row, int hi) {
#pragma unroll
    for (int s = 0; s < 2; ++s) {
        int j = wave * 32 + s * 16 + row;
#pragma unroll
        for (int v = 0; v < 8; ++v) {
            int b = v + 8 * hi;
            float iv = sigf(acc[0][s][v] + bs[0][s]);
            float fv = sigf(acc[1][s][v] + bs[1][s]);
            float gv = tanhf(acc[2][s][v] + bs[2][s]);
            float ov = sigf(acc[3][s][v] + bs[3][s]);
            float cc = fv * creg[s][v] + iv * gv;
            creg[s][v] = cc;
            hbuf[b * 256 + j] = (_Float16)(ov * tanhf(cc));
        }
    }
}

// ---------------------------------------------------------------- full seq model: encoder, attention, decoder, heads
__global__ __launch_bounds__(256) void lstm_seq_kernel(
        const float* __restrict__ diffs,     // [16][10][6]
        const float* __restrict__ bf,        // [160][1024]
        const _Float16* __restrict__ eWih,   // [1024][1056] f16 (K padded)
        const _Float16* __restrict__ eWhh,   // [1024][256]
        const float* __restrict__ enc_bih, const float* __restrict__ enc_bhh,
        const float* __restrict__ attn_w,    // [10][1286]
        const float* __restrict__ attn_b,
        const _Float16* __restrict__ dWih,   // [1024][1312] f16 (K padded)
        const _Float16* __restrict__ dWhh,   // [1024][256]
        const float* __restrict__ dec_bih, const float* __restrict__ dec_bhh,
        const float* __restrict__ lin1_w, const float* __restrict__ lin1_b,
        const float* __restrict__ lin2_w, const float* __restrict__ lin2_b,
        float* __restrict__ enc_out,         // ws [10][16][256]
        float* __restrict__ out)             // [16][6]
{
    __shared__ __attribute__((aligned(16))) _Float16 xbuf[16 * 1312];
    __shared__ __attribute__((aligned(16))) _Float16 hbuf[16 * 256];
    __shared__ __attribute__((aligned(16))) _Float16 obuf[16 * 256];
    __shared__ float aw[160];

    const int tid = threadIdx.x;
    const int wave = tid >> 5, lane = tid & 31;
    const int row = lane & 15, hi = lane >> 4;

    for (int i = tid; i < 16 * 256; i += 256) hbuf[i] = (_Float16)0.f;

    float creg[2][8];
#pragma unroll
    for (int s = 0; s < 2; ++s)
#pragma unroll
        for (int v = 0; v < 8; ++v) creg[s][v] = 0.f;

    float ebs[4][2], dbs[4][2];
#pragma unroll
    for (int g = 0; g < 4; ++g)
#pragma unroll
        for (int s = 0; s < 2; ++s) {
            int n = g * 256 + wave * 32 + s * 16 + row;
            ebs[g][s] = enc_bih[n] + enc_bhh[n];
            dbs[g][s] = dec_bih[n] + dec_bhh[n];
        }
    __syncthreads();

    // -------- encoder: 10 steps
    for (int t = 0; t < 10; ++t) {
        for (int idx = tid; idx < 16 * 1312; idx += 256) {
            int b = idx / 1312, k = idx - b * 1312;
            float v = 0.f;
            if (k < 6) v = diffs[(b * 10 + t) * 6 + k];
            else if (k < 1030 && t > 0) v = bf[(b * 10 + (t - 1)) * 1024 + (k - 6)];
            xbuf[idx] = (_Float16)v;
        }
        __syncthreads();

        f32x8 acc[4][2];
        f32x8 z = {0.f, 0.f, 0.f, 0.f, 0.f, 0.f, 0.f, 0.f};
#pragma unroll
        for (int g = 0; g < 4; ++g)
#pragma unroll
            for (int s = 0; s < 2; ++s) acc[g][s] = z;

        gate_gemm(xbuf, 1056, eWih, hbuf, eWhh, wave, row, hi, acc);
        __syncthreads();                       // all waves done reading hbuf
        lstm_combine(acc, ebs, creg, hbuf, wave, row, hi);
        __syncthreads();
        for (int idx = tid; idx < 4096; idx += 256)
            enc_out[t * 4096 + idx] = (float)hbuf[idx];
        __threadfence_block();
        __syncthreads();
    }

    // -------- decoder input dec0 = [0,0,0,0,1,0, bf[:,9]]
    for (int idx = tid; idx < 16 * 1312; idx += 256) {
        int b = idx / 1312, k = idx - b * 1312;
        float v = 0.f;
        if (k == 4) v = 1.f;
        else if (k >= 6 && k < 1030) v = bf[(b * 10 + 9) * 1024 + (k - 6)];
        xbuf[idx] = (_Float16)v;
    }
    __syncthreads();

    // -------- attention: softmax((concat(h, dec0) @ attn_w.T) + attn_b)
    if (tid < 160) {
        int b = tid / 10, l = tid - b * 10;
        const float* wrow = attn_w + l * 1286;
        float s = attn_b[l];
        for (int k = 0; k < 256; ++k) s += wrow[k] * (float)hbuf[b * 256 + k];
        for (int k = 0; k < 1030; ++k) s += wrow[256 + k] * (float)xbuf[b * 1312 + k];
        aw[b * 10 + l] = s;
    }
    __syncthreads();
    if (tid < 16) {
        float m = aw[tid * 10];
        for (int l = 1; l < 10; ++l) m = fmaxf(m, aw[tid * 10 + l]);
        float e[10], sum = 0.f;
        for (int l = 0; l < 10; ++l) { e[l] = __expf(aw[tid * 10 + l] - m); sum += e[l]; }
        for (int l = 0; l < 10; ++l) aw[tid * 10 + l] = e[l] / sum;
    }
    __syncthreads();
    // applied = aw @ enc_out  -> xbuf[:, 1030:1286]
    for (int idx = tid; idx < 4096; idx += 256) {
        int b = idx >> 8, j = idx & 255;
        float s = 0.f;
        for (int l = 0; l < 10; ++l) s += aw[b * 10 + l] * enc_out[(l * 16 + b) * 256 + j];
        xbuf[b * 1312 + 1030 + j] = (_Float16)s;
    }
    __syncthreads();

    // -------- decoder LSTM step (c starts at 0)
#pragma unroll
    for (int s = 0; s < 2; ++s)
#pragma unroll
        for (int v = 0; v < 8; ++v) creg[s][v] = 0.f;
    {
        f32x8 acc[4][2];
        f32x8 z = {0.f, 0.f, 0.f, 0.f, 0.f, 0.f, 0.f, 0.f};
#pragma unroll
        for (int g = 0; g < 4; ++g)
#pragma unroll
            for (int s = 0; s < 2; ++s) acc[g][s] = z;
        gate_gemm(xbuf, 1312, dWih, hbuf, dWhh, wave, row, hi, acc);
        __syncthreads();
        lstm_combine(acc, dbs, creg, hbuf, wave, row, hi);
        __syncthreads();
    }

    // -------- heads: relu(h @ lin1.T + b1) @ lin2.T + b2
    for (int idx = tid; idx < 4096; idx += 256) {
        int b = idx >> 8, j = idx & 255;
        float s = lin1_b[j];
        const float* wr = lin1_w + j * 256;
        for (int k = 0; k < 256; ++k) s += wr[k] * (float)hbuf[b * 256 + k];
        obuf[idx] = (_Float16)fmaxf(s, 0.f);
    }
    __syncthreads();
    if (tid < 96) {
        int b = tid / 6, o = tid - b * 6;
        float s = lin2_b[o];
        const float* wr = lin2_w + o * 256;
        for (int k = 0; k < 256; ++k) s += wr[k] * (float)obuf[b * 256 + k];
        out[b * 6 + o] = s;
    }
}

// ---------------------------------------------------------------- launcher
extern "C" void kernel_launch(void* const* d_in, const int* in_sizes, int n_in,
                              void* d_out, int out_size, void* d_ws, size_t ws_size,
                              hipStream_t stream) {
    (void)in_sizes; (void)n_in; (void)out_size; (void)ws_size;

    const float* diffs          = (const float*)d_in[0];
    const float* boxes_resized  = (const float*)d_in[2];
    const float* image_features = (const float*)d_in[3];
    const int*   image_sizes    = (const int*)d_in[4];
    const float* conv3_1_w = (const float*)d_in[6];
    const float* conv3_1_b = (const float*)d_in[7];
    const float* conv4_w   = (const float*)d_in[8];
    const float* conv4_b   = (const float*)d_in[9];
    const float* conv4_1_w = (const float*)d_in[10];
    const float* conv4_1_b = (const float*)d_in[11];
    const float* enc_Wih = (const float*)d_in[12];
    const float* enc_Whh = (const float*)d_in[13];
    const float* enc_bih = (const float*)d_in[14];
    const float* enc_bhh = (const float*)d_in[15];
    const float* attn_w  = (const float*)d_in[16];
    const float* attn_b  = (const float*)d_in[17];
    const float* dec_Wih = (const float*)d_in[18];
    const float* dec_Whh = (const float*)d_in[19];
    const float* dec_bih = (const float*)d_in[20];
    const float* dec_bhh = (const float*)d_in[21];
    const float* lin1_w = (const float*)d_in[22];
    const float* lin1_b = (const float*)d_in[23];
    const float* lin2_w = (const float*)d_in[24];
    const float* lin2_b = (const float*)d_in[25];
    float* out = (float*)d_out;

    char* ws = (char*)d_ws;
    size_t off = 0;
    auto alloc = [&](size_t bytes) -> char* {
        char* p = ws + off;
        off = (off + bytes + 255) & ~(size_t)255;
        return p;
    };
    _Float16* regionA = (_Float16*)alloc((size_t)NPAIR * 128 * HW * 2); // corr / conv4 out
    _Float16* regionB = (_Float16*)alloc((size_t)NPAIR * 128 * HW * 2); // conv3 / conv4_1 out
    _Float16* w3   = (_Float16*)alloc((size_t)128 * 736 * 2);
    _Float16* w4   = (_Float16*)alloc((size_t)128 * 1152 * 2);
    _Float16* w41  = (_Float16*)alloc((size_t)64 * 1152 * 2);
    _Float16* eWih16 = (_Float16*)alloc((size_t)1024 * 1056 * 2);
    _Float16* eWhh16 = (_Float16*)alloc((size_t)1024 * 256 * 2);
    _Float16* dWih16 = (_Float16*)alloc((size_t)1024 * 1312 * 2);
    _Float16* dWhh16 = (_Float16*)alloc((size_t)1024 * 256 * 2);
    float* bf      = (float*)alloc((size_t)NPAIR * 1024 * 4);
    float* enc_out = (float*)alloc((size_t)10 * 16 * 256 * 4);

    auto cvt = [&](const float* s, _Float16* d, int O, int K, int Kpad) {
        int total = O * Kpad;
        wcvt_kernel<<<(total + 255) / 256, 256, 0, stream>>>(s, d, K, Kpad, total);
    };
    cvt(conv3_1_w, w3, 128, 729, 736);
    cvt(conv4_w, w4, 128, 1152, 1152);
    cvt(conv4_1_w, w41, 64, 1152, 1152);
    cvt(enc_Wih, eWih16, 1024, 1030, 1056);
    cvt(enc_Whh, eWhh16, 1024, 256, 256);
    cvt(dec_Wih, dWih16, 1024, 1286, 1312);
    cvt(dec_Whh, dWhh16, 1024, 256, 256);

    // correlation -> regionA (f16, [160][81][1008])
    corr_kernel<<<(NPAIR * 81 * HW) / 256, 256, 0, stream>>>(image_features, regionA);

    // conv3_1: 81 -> 128, conv4: 128 -> 128, conv4_1: 128 -> 64 (all leaky 0.1)
    // grid.x = 1008/48 = 21 position tiles, one wave per 16-Cout tile
    conv_wmma_kernel<<<dim3(21, NPAIR), 256, 0, stream>>>(regionA, w3, conv3_1_b, regionB, 81, 128, 736);
    conv_wmma_kernel<<<dim3(21, NPAIR), 256, 0, stream>>>(regionB, w4, conv4_b, regionA, 128, 128, 1152);
    conv_wmma_kernel<<<dim3(21, NPAIR), 128, 0, stream>>>(regionA, w41, conv4_1_b, regionB, 128, 64, 1152);

    // roi align -> bf [160][1024]
    roi_kernel<<<(NPAIR * 64 * 16) / 256, 256, 0, stream>>>(regionB, boxes_resized, image_sizes, bf);

    // full recurrent tail in one workgroup
    lstm_seq_kernel<<<1, 256, 0, stream>>>(diffs, bf, eWih16, eWhh16, enc_bih, enc_bhh,
                                           attn_w, attn_b, dWih16, dWhh16, dec_bih, dec_bhh,
                                           lin1_w, lin1_b, lin2_w, lin2_b, enc_out, out);
}
